// MultiHeadAttention_62483184222306
// MI455X (gfx1250) — matrix operand order repaired and checked by reading, not compiled, not run on hardware
//
#include <hip/hip_runtime.h>
#include <hip/hip_bf16.h>
#include <stdint.h>

#define EMBED  512
#define HEADS  8
#define HD     64
#define NBATCH 4
#define SEQ    2048

typedef __attribute__((ext_vector_type(16))) __bf16 bf16x16;
typedef __attribute__((ext_vector_type(2)))  __bf16 bf16x2;
typedef __attribute__((ext_vector_type(8)))  float  v8f;

union FragU { uint4 q[2]; bf16x16 f; };

// ---- fp32 -> bf16 pair conversion -------------------------------------
// CDNA5 VOP3P mixed-precision converts (ISA 15.10 opcodes 62/63):
//   D[15:0]/[31:16].bf16 = cvt(fma(S0, 1.0, 0.0)) with f32 sources.
#if __has_builtin(__builtin_amdgcn_cvt_pk_bf16_f32)
__device__ __forceinline__ unsigned int pack2(float a, float b) {
  union { bf16x2 v; unsigned int u; } c;
  c.v = __builtin_amdgcn_cvt_pk_bf16_f32(a, b);   // v_cvt_pk_bf16_f32
  return c.u;
}
#elif defined(__gfx1250__)
__device__ __forceinline__ unsigned int pack2(float a, float b) {
  unsigned int d;
  asm("v_fma_mixlo_bf16 %0, %1, 1.0, 0\n\t"
      "v_fma_mixhi_bf16 %0, %2, 1.0, 0"
      : "=v"(d)
      : "v"(a), "v"(b));
  return d;
}
#else
__device__ __forceinline__ unsigned int pack2(float a, float b) {
  unsigned int ua = __float_as_uint(a);
  ua += 0x7fffu + ((ua >> 16) & 1u);
  unsigned int ub = __float_as_uint(b);
  ub += 0x7fffu + ((ub >> 16) & 1u);
  return (ua >> 16) | (ub & 0xffff0000u);
}
#endif

#if __has_builtin(__builtin_amdgcn_exp2f)
#define EXP2F(x) __builtin_amdgcn_exp2f(x)        // v_exp_f32
#else
#define EXP2F(x) exp2f(x)
#endif

// Same-wave LDS store->load ordering: LDS is in-order per wave on CDNA5, so a
// wavefront-scope fence (compiler ordering only, no s_barrier) is sufficient
// when each wave touches only its own LDS region.
#if __has_builtin(__builtin_amdgcn_fence)
#define WAVE_LDS_FENCE() __builtin_amdgcn_fence(__ATOMIC_ACQ_REL, "wavefront")
#else
#define WAVE_LDS_FENCE() __syncthreads()
#endif

// A-fragment (16x32 bf16, ISA 7.12.2): lane holds row (lane&15);
// elems 0..7 -> K = kb + 8*half + j ; elems 8..15 -> K = kb + 16 + 8*half + j
__device__ __forceinline__ bf16x16 load_a(const unsigned short* row, int kb, int half) {
  FragU u;
  u.q[0] = *(const uint4*)(row + kb + 8 * half);
  u.q[1] = *(const uint4*)(row + kb + 16 + 8 * half);
  return u.f;
}
// B-fragment (32x16 bf16): lane holds column (lane&15) == source row of the
// transposed operand; elems 0..7 -> K = kb + 8*half + j ; elems 8..15 -> K = kb + 16 + 8*half + j
__device__ __forceinline__ bf16x16 load_b(const unsigned short* row, int kb, int half) {
  FragU u;
  u.q[0] = *(const uint4*)(row + kb + 8 * half);
  u.q[1] = *(const uint4*)(row + kb + 16 + 8 * half);
  return u.f;
}
// A-fragment loaded from fp32 memory with on-the-fly packed bf16 downconvert.
__device__ __forceinline__ bf16x16 load_a_f32(const float* row, int kb, int half) {
  FragU u;
  const float4* c0 = (const float4*)(row + kb + 8 * half);
  const float4* c1 = (const float4*)(row + kb + 16 + 8 * half);
  float4 a = c0[0], b = c0[1];
  u.q[0] = make_uint4(pack2(a.x, a.y), pack2(a.z, a.w), pack2(b.x, b.y), pack2(b.z, b.w));
  a = c1[0]; b = c1[1];
  u.q[1] = make_uint4(pack2(a.x, a.y), pack2(a.z, a.w), pack2(b.x, b.y), pack2(b.z, b.w));
  return u.f;
}

__device__ __forceinline__ v8f wmma_bf16(bf16x16 a, bf16x16 b, v8f c) {
  return __builtin_amdgcn_wmma_f32_16x16x32_bf16(false, a, false, b, (short)0, c,
                                                 false, false);
}

// ---------------------------------------------------------------- stage 1
// Packed fp32 -> bf16 downconvert (element counts here are all even).
__global__ void cvt_bf16_kernel(const float* __restrict__ src,
                                unsigned int* __restrict__ dst, int npairs) {
  int i = blockIdx.x * blockDim.x + threadIdx.x;
  if (i < npairs) {
    float2 v = ((const float2*)src)[i];
    dst[i] = pack2(v.x, v.y);
  }
}

// ---------------------------------------------------------------- stage 2
// Per-head projection  y = x_head @ W^T  (x fp32 -> bf16 fragments).
// sel 0: query->qp (row major)   1: keys->kp (row major)   2: values->vpt (d-major)
__global__ __launch_bounds__(128) void project_kernel(
    const float* __restrict__ q_in, const float* __restrict__ k_in,
    const float* __restrict__ v_in, const unsigned short* __restrict__ Wq,
    const unsigned short* __restrict__ Wk, const unsigned short* __restrict__ Wv,
    unsigned short* __restrict__ qp, unsigned short* __restrict__ kp,
    unsigned short* __restrict__ vpt) {
  const int sel  = blockIdx.y;
  const int wave = threadIdx.x >> 5;
  const int lane = threadIdx.x & 31;
  const int half = lane >> 4, l15 = lane & 15;

  const float*          x = (sel == 0) ? q_in : (sel == 1) ? k_in : v_in;
  const unsigned short* W = (sel == 0) ? Wq   : (sel == 1) ? Wk   : Wv;

  const int tile = blockIdx.x * 4 + wave;          // 0 .. N*H*(S/16)-1
  const int nh   = tile / (SEQ / 16);
  const int s0   = (tile % (SEQ / 16)) * 16;
  const int n    = nh / HEADS, h = nh % HEADS;

  // Preload all W fragments first (one clause, loads in flight during cvt).
  bf16x16 wb[4][2];
#pragma unroll
  for (int dt = 0; dt < 4; ++dt) {
    const unsigned short* wrow = W + (size_t)(dt * 16 + l15) * HD;  // W[dout][din]
    wb[dt][0] = load_b(wrow, 0, half);
    wb[dt][1] = load_b(wrow, 32, half);
  }

  const float* xrow = x + ((size_t)n * SEQ + s0 + l15) * EMBED + h * HD;
  bf16x16 a0 = load_a_f32(xrow, 0, half);
  bf16x16 a1 = load_a_f32(xrow, 32, half);

  v8f acc[4] = {};
#pragma unroll
  for (int dt = 0; dt < 4; ++dt) {
    acc[dt] = wmma_bf16(a0, wb[dt][0], acc[dt]);
    acc[dt] = wmma_bf16(a1, wb[dt][1], acc[dt]);
  }

  if (sel < 2) {
    unsigned short* out = (sel == 0) ? qp : kp;
    unsigned short* orow0 = out + ((size_t)nh * SEQ + s0 + 8 * half) * HD;
#pragma unroll
    for (int r = 0; r < 8; ++r) {
      unsigned short* orow = orow0 + (size_t)r * HD;
      unsigned int u01 = pack2(acc[0][r], acc[1][r]);
      unsigned int u23 = pack2(acc[2][r], acc[3][r]);
      orow[l15]      = (unsigned short)u01;
      orow[16 + l15] = (unsigned short)(u01 >> 16);
      orow[32 + l15] = (unsigned short)u23;
      orow[48 + l15] = (unsigned short)(u23 >> 16);
    }
  } else {
#pragma unroll
    for (int dt = 0; dt < 4; ++dt) {
      unsigned short* ocol =
          vpt + ((size_t)nh * HD + dt * 16 + l15) * SEQ + s0 + 8 * half;
#pragma unroll
      for (int r = 0; r < 8; r += 2) {
        unsigned int u = pack2(acc[dt][r], acc[dt][r + 1]);
        ocol[r]     = (unsigned short)u;        // consecutive s -> contiguous
        ocol[r + 1] = (unsigned short)(u >> 16);
      }
    }
  }
}

// ---------------------------------------------------------------- stage 3
// Flash attention: one wave per 16-row Q tile; kv chunks of 32.
// Softmax runs in the exp2 domain: energy scale = log2(e)/sqrt(d).
__global__ __launch_bounds__(128) void flash_attn_kernel(
    const unsigned short* __restrict__ qp, const unsigned short* __restrict__ kp,
    const unsigned short* __restrict__ vpt, const int* __restrict__ mask,
    unsigned short* __restrict__ attn_out) {
  __shared__ unsigned short smP[4][16 * 32];       // per-wave P staging tile

  const int wave = threadIdx.x >> 5;
  const int lane = threadIdx.x & 31;
  const int half = lane >> 4, l15 = lane & 15;

  const int tile = blockIdx.x * 4 + wave;          // 0 .. N*H*(S/16)-1
  const int nh   = tile / (SEQ / 16);
  const int q0   = (tile % (SEQ / 16)) * 16;
  const int n    = nh / HEADS, h = nh % HEADS;

  const unsigned short* qrow = qp + ((size_t)nh * SEQ + q0 + l15) * HD;
  const bf16x16 qa0 = load_a(qrow, 0, half);       // Q resident in registers
  const bf16x16 qa1 = load_a(qrow, 32, half);

  const unsigned short* kbase = kp + (size_t)nh * SEQ * HD;
  const unsigned short* vhead = vpt + (size_t)nh * HD * SEQ;
  const int*            mrow  = mask + (size_t)n * SEQ;

  const float c_scale = 0.125f * 1.44269504088896341f;  // log2e / sqrt(64)

  v8f acc0 = {}, acc1 = {}, acc2 = {}, acc3 = {};
  float M[8], L[8];
#pragma unroll
  for (int r = 0; r < 8; ++r) { M[r] = -3.0e38f; L[r] = 0.0f; }

#pragma unroll 2
  for (int k0 = 0; k0 < SEQ; k0 += 32) {
    const unsigned short* krowA = kbase + (size_t)(k0 + l15) * HD;
    const unsigned short* krowB = krowA + 16 * HD;
    if (k0 + 32 < SEQ) {                            // global_prefetch next tiles
      __builtin_prefetch(krowA + 32 * HD, 0, 0);
      __builtin_prefetch(vhead + (size_t)l15 * SEQ + k0 + 32, 0, 0);
    }
    // Preload all K fragments (distinct regs -> one load clause in flight).
    const bf16x16 kb0 = load_b(krowA, 0, half);
    const bf16x16 kb1 = load_b(krowA, 32, half);
    const bf16x16 kb2 = load_b(krowB, 0, half);
    const bf16x16 kb3 = load_b(krowB, 32, half);
    v8f e0 = {}, e1 = {};
    e0 = wmma_bf16(qa0, kb0, e0);
    e0 = wmma_bf16(qa1, kb1, e0);
    e1 = wmma_bf16(qa0, kb2, e1);
    e1 = wmma_bf16(qa1, kb3, e1);

    // Issue V-fragment loads now: their latency hides under the softmax VALU.
    const bf16x16 vb0 = load_b(vhead + (size_t)(l15)      * SEQ + k0, 0, half);
    const bf16x16 vb1 = load_b(vhead + (size_t)(16 + l15) * SEQ + k0, 0, half);
    const bf16x16 vb2 = load_b(vhead + (size_t)(32 + l15) * SEQ + k0, 0, half);
    const bf16x16 vb3 = load_b(vhead + (size_t)(48 + l15) * SEQ + k0, 0, half);

    const int m0 = mrow[k0 + l15];                  // column of e0 for this lane
    const int m1 = mrow[k0 + 16 + l15];             // column of e1

#pragma unroll
    for (int r = 0; r < 8; ++r) {
      float x0 = e0[r] * c_scale; if (m0 == 0) x0 = -1.0e20f;
      float x1 = e1[r] * c_scale; if (m1 == 0) x1 = -1.0e20f;
      float t = fmaxf(x0, x1);                      // row max over 16-lane group
      t = fmaxf(t, __shfl_xor(t, 1, 32));
      t = fmaxf(t, __shfl_xor(t, 2, 32));
      t = fmaxf(t, __shfl_xor(t, 4, 32));
      t = fmaxf(t, __shfl_xor(t, 8, 32));
      const float Mn    = fmaxf(M[r], t);
      const float alpha = EXP2F(M[r] - Mn);
      const float p0 = EXP2F(x0 - Mn), p1 = EXP2F(x1 - Mn);
      float s = p0 + p1;                            // row sum
      s += __shfl_xor(s, 1, 32);
      s += __shfl_xor(s, 2, 32);
      s += __shfl_xor(s, 4, 32);
      s += __shfl_xor(s, 8, 32);
      L[r] = L[r] * alpha + s;
      M[r] = Mn;
      acc0[r] *= alpha; acc1[r] *= alpha; acc2[r] *= alpha; acc3[r] *= alpha;
      const unsigned int pu = pack2(p0, p1);        // one packed cvt per row
      smP[wave][(r + 8 * half) * 32 + l15]      = (unsigned short)pu;
      smP[wave][(r + 8 * half) * 32 + 16 + l15] = (unsigned short)(pu >> 16);
    }
    WAVE_LDS_FENCE();                // C-layout -> A-layout via per-wave LDS;
    const bf16x16 pa = load_a(&smP[wave][l15 * 32], 0, half);
    acc0 = wmma_bf16(pa, vb0, acc0);
    acc1 = wmma_bf16(pa, vb1, acc1);
    acc2 = wmma_bf16(pa, vb2, acc2);
    acc3 = wmma_bf16(pa, vb3, acc3);
  }

#pragma unroll
  for (int r = 0; r < 8; ++r) {
    const float inv = 1.0f / L[r];
    unsigned short* orow =
        attn_out + ((size_t)n * SEQ + q0 + r + 8 * half) * EMBED + h * HD;
    const unsigned int u01 = pack2(acc0[r] * inv, acc1[r] * inv);
    const unsigned int u23 = pack2(acc2[r] * inv, acc3[r] * inv);
    orow[l15]      = (unsigned short)u01;
    orow[16 + l15] = (unsigned short)(u01 >> 16);
    orow[32 + l15] = (unsigned short)u23;
    orow[48 + l15] = (unsigned short)(u23 >> 16);
  }
}

// ---------------------------------------------------------------- stage 4
// out = attn_bf16 @ Wo^T + bo   ->  fp32 d_out.  K-loop unrolled x2 with all
// fragments preloaded so loads stay in flight across the WMMAs.
__global__ __launch_bounds__(128) void out_proj_kernel(
    const unsigned short* __restrict__ xin, const unsigned short* __restrict__ Wo,
    const float* __restrict__ bo, float* __restrict__ out) {
  const int wave = threadIdx.x >> 5;
  const int lane = threadIdx.x & 31;
  const int half = lane >> 4, l15 = lane & 15;

  const int tile = blockIdx.x * 4 + wave;          // 512 row-tiles x 8 col-groups
  const int r0 = (tile >> 3) * 16;
  const int c0 = (tile & 7) * 64;

  const unsigned short* xrow = xin + (size_t)(r0 + l15) * EMBED;
  v8f acc[4] = {};
  for (int k = 0; k < EMBED; k += 64) {
    const bf16x16 a0 = load_a(xrow, k, half);
    const bf16x16 a1 = load_a(xrow, k + 32, half);
    bf16x16 b0[4], b1[4];
#pragma unroll
    for (int dt = 0; dt < 4; ++dt) {
      const unsigned short* wrow = Wo + (size_t)(c0 + dt * 16 + l15) * EMBED;
      b0[dt] = load_b(wrow, k, half);
      b1[dt] = load_b(wrow, k + 32, half);
    }
#pragma unroll
    for (int dt = 0; dt < 4; ++dt) {
      acc[dt] = wmma_bf16(a0, b0[dt], acc[dt]);
      acc[dt] = wmma_bf16(a1, b1[dt], acc[dt]);
    }
  }
#pragma unroll
  for (int dt = 0; dt < 4; ++dt) {
    const int   c = c0 + dt * 16 + l15;
    const float b = bo[c];
#pragma unroll
    for (int r = 0; r < 8; ++r)
      out[(size_t)(r0 + r + 8 * half) * EMBED + c] = acc[dt][r] + b;
  }
}

// ---------------------------------------------------------------- launch
extern "C" void kernel_launch(void* const* d_in, const int* in_sizes, int n_in,
                              void* d_out, int out_size, void* d_ws, size_t ws_size,
                              hipStream_t stream) {
  const float* values = (const float*)d_in[0];
  const float* keys   = (const float*)d_in[1];
  const float* query  = (const float*)d_in[2];
  const int*   mask   = (const int*)d_in[3];
  const float* Wv     = (const float*)d_in[4];
  const float* Wk     = (const float*)d_in[5];
  const float* Wq     = (const float*)d_in[6];
  const float* Wo     = (const float*)d_in[7];
  const float* bo     = (const float*)d_in[8];
  float*       out    = (float*)d_out;

  unsigned short* p = (unsigned short*)d_ws;
  unsigned short* Wq_bf = p;  p += 64 * 64;
  unsigned short* Wk_bf = p;  p += 64 * 64;
  unsigned short* Wv_bf = p;  p += 64 * 64;
  unsigned short* Wo_bf = p;  p += EMBED * EMBED;
  unsigned short* qproj = p;  p += (size_t)NBATCH * HEADS * SEQ * HD;
  unsigned short* kproj = p;  p += (size_t)NBATCH * HEADS * SEQ * HD;
  unsigned short* vptT  = p;  p += (size_t)NBATCH * HEADS * HD * SEQ;
  unsigned short* attn  = p;  // N*S*EMBED bf16

  cvt_bf16_kernel<<<(64 * 64 / 2 + 255) / 256, 256, 0, stream>>>(
      Wq, (unsigned int*)Wq_bf, 64 * 64 / 2);
  cvt_bf16_kernel<<<(64 * 64 / 2 + 255) / 256, 256, 0, stream>>>(
      Wk, (unsigned int*)Wk_bf, 64 * 64 / 2);
  cvt_bf16_kernel<<<(64 * 64 / 2 + 255) / 256, 256, 0, stream>>>(
      Wv, (unsigned int*)Wv_bf, 64 * 64 / 2);
  cvt_bf16_kernel<<<(EMBED * EMBED / 2 + 255) / 256, 256, 0, stream>>>(
      Wo, (unsigned int*)Wo_bf, EMBED * EMBED / 2);

  const int qtiles = NBATCH * HEADS * (SEQ / 16);   // 4096 waves, 4 per block
  project_kernel<<<dim3(qtiles / 4, 3), 128, 0, stream>>>(
      query, keys, values, Wq_bf, Wk_bf, Wv_bf, qproj, kproj, vptT);

  flash_attn_kernel<<<qtiles / 4, 128, 0, stream>>>(qproj, kproj, vptT, mask, attn);

  const int otiles = (NBATCH * SEQ / 16) * (EMBED / 64);  // 4096 waves
  out_proj_kernel<<<otiles / 4, 128, 0, stream>>>(attn, Wo_bf, bo, out);
}
